// SelfAttention_53807350284603
// MI455X (gfx1250) — compile-verified, matrix operand order checked by reading, not run on hardware
//
#include <hip/hip_runtime.h>
#include <hip/hip_bf16.h>
#include <cmath>

// ---------------- problem constants ----------------
#define B_ 8
#define C_ 512
#define L_ 2048
#define GROUPS_ 8
#define CG_ (C_ / GROUPS_)   // 64
#define EPS_ 1e-5f

typedef __attribute__((ext_vector_type(16))) __bf16 v16bf;
typedef __attribute__((ext_vector_type(8)))  float  v8f;

__device__ __forceinline__ unsigned short f2bf(float f) {
    unsigned u = __float_as_uint(f);
    unsigned r = u + 0x7FFFu + ((u >> 16) & 1u);   // round-to-nearest-even
    return (unsigned short)(r >> 16);
}

// ---------------- f32 -> bf16 convert ----------------
__global__ __launch_bounds__(256) void f32_to_bf16_kernel(
    const float* __restrict__ in, unsigned short* __restrict__ out, long n) {
    long i = (long)blockIdx.x * 256 + threadIdx.x;
    if (i < n) out[i] = f2bf(in[i]);
}

// ---------------- GroupNorm (f32 in, bf16 out) ----------------
__global__ __launch_bounds__(256) void groupnorm_bf16_kernel(
    const float* __restrict__ x, const float* __restrict__ gamma,
    const float* __restrict__ beta, unsigned short* __restrict__ xn) {
    const int b = blockIdx.y;
    const int g = blockIdx.x;
    const int tid = threadIdx.x;
    __shared__ float sSum[256], sSq[256];
    const long base = ((long)b * C_ + (long)g * CG_) * L_;
    const int n = CG_ * L_;                      // 131072
    float s = 0.f, ss = 0.f;
    for (int i = tid; i < n; i += 256) {
        float v = x[base + i];
        s += v; ss += v * v;
    }
    sSum[tid] = s; sSq[tid] = ss;
    __syncthreads();
    for (int off = 128; off > 0; off >>= 1) {
        if (tid < off) { sSum[tid] += sSum[tid + off]; sSq[tid] += sSq[tid + off]; }
        __syncthreads();
    }
    const float mu   = sSum[0] / (float)n;
    const float var  = sSq[0] / (float)n - mu * mu;
    const float rstd = rsqrtf(var + EPS_);
    for (int i = tid; i < n; i += 256) {
        int c = g * CG_ + (i >> 11);             // i / L_
        float v = (x[base + i] - mu) * rstd * gamma[c] + beta[c];
        xn[base + i] = f2bf(v);
    }
}

// ---------------- bf16 WMMA GEMM ----------------
// C[M,N] = A[M,K] * B[K,N].  A must be k-contiguous (sAk==1).
// BMODE 0: B is k-contiguous (sBk==1)  -> async b128 copy straight to LDS
// BMODE 1: B is n-contiguous (sBn==1)  -> b128 global loads + LDS transpose scatter
#define BM 128
#define BN 256
#define BK 32
#define LDA_ 40   // padded K-stride in LDS (bf16 elems)
#define LDB_ 40

enum { EPI_BF16 = 0, EPI_F32_SCALE = 1, EPI_F32_BIAS_RES = 2, EPI_QKV = 3 };

union FragAB { v16bf v; unsigned u[8]; };
union BStage { uint4 v; unsigned short h[8]; };

__device__ __forceinline__ unsigned lds_off32(const void* p) {
    return (unsigned)(unsigned long long)p;      // low 32 bits of generic addr = LDS offset
}

__device__ __forceinline__ void async_b128(unsigned loff, unsigned goff,
                                           const unsigned short* sbase) {
    asm volatile("global_load_async_to_lds_b128 %0, %1, %2"
                 :: "v"(loff), "v"(goff), "s"(sbase) : "memory");
}

template <int EPI, int BMODE>
__global__ __launch_bounds__(256) void gemm_bf16_wmma(
    const unsigned short* __restrict__ A, const unsigned short* __restrict__ Bm,
    long sAm, long sBk, long sBn,
    long batchA, long batchB, long batchC, long batchR,
    int M, int N, int K,
    const float* __restrict__ bias, const float* __restrict__ resid,
    float scale, void* __restrict__ Cout, int ldC,
    unsigned short* __restrict__ aux)            // qT output for EPI_QKV
{
    __shared__ unsigned short lhs[2][BM * LDA_];   // 2 x 10240 B
    __shared__ unsigned short rhs[2][BN * LDB_];   // 2 x 20480 B

    const int tid = threadIdx.x;
    const int bat = blockIdx.y;
    const int ntN = N / BN;
    const int m0 = (blockIdx.x / ntN) * BM;
    const int n0 = (blockIdx.x % ntN) * BN;

    const unsigned short* Ab = A  + (long)bat * batchA;
    const unsigned short* Bb = Bm + (long)bat * batchB;

    const int wave = tid >> 5;
    const int lane = tid & 31;
    const int lr   = lane & 15;
    const int hs   = lane >> 4;                  // half-select per ISA 16b layout
    const int mW   = (wave >> 2) * 64;           // 2 waves along M
    const int nW   = (wave & 3) * 64;            // 4 waves along N

    // ---- staging helpers ----
    // A tile: BM*BK = 4096 elems, 8 per slot -> 512 slots, 2 per thread (async b128)
    auto issueA = [&](int k0, int buf) {
        #pragma unroll
        for (int t = 0; t < 2; t++) {
            const int slot = tid + t * 256;
            const int m = slot >> 2, k8 = (slot & 3) * 8;
            const unsigned loff = lds_off32(&lhs[buf][m * LDA_ + k8]);
            const unsigned goff = (unsigned)((((long)(m0 + m)) * sAm + (k0 + k8)) * 2);
            async_b128(loff, goff, Ab);
        }
    };
    // B tile BMODE 0: BN*BK = 8192 elems -> 1024 slots, 4 per thread (async b128)
    auto issueB0 = [&](int k0, int buf) {
        #pragma unroll
        for (int t = 0; t < 4; t++) {
            const int slot = tid + t * 256;
            const int n = slot >> 2, k8 = (slot & 3) * 8;
            const unsigned loff = lds_off32(&rhs[buf][n * LDB_ + k8]);
            const unsigned goff = (unsigned)((((long)(n0 + n)) * sBn + (k0 + k8)) * 2);
            async_b128(loff, goff, Bb);
        }
    };
    // B tile BMODE 1: b128 global loads (n-contiguous) into regs
    BStage bregs[4];
    auto loadB1 = [&](int k0) {
        #pragma unroll
        for (int t = 0; t < 4; t++) {
            const int slot = tid + t * 256;
            const int k = slot >> 5, n8 = (slot & 31) * 8;
            bregs[t].v = *(const uint4*)(Bb + (long)(k0 + k) * sBk + (n0 + n8));
        }
    };
    auto commitB1 = [&](int buf) {
        #pragma unroll
        for (int t = 0; t < 4; t++) {
            const int slot = tid + t * 256;
            const int k = slot >> 5, n8 = (slot & 31) * 8;
            #pragma unroll
            for (int j = 0; j < 8; j++)
                rhs[buf][(n8 + j) * LDB_ + k] = bregs[t].h[j];
        }
    };

    const v8f vzero = {0.f,0.f,0.f,0.f,0.f,0.f,0.f,0.f};
    v8f acc[4][4];
    #pragma unroll
    for (int i = 0; i < 4; i++)
        #pragma unroll
        for (int j = 0; j < 4; j++) acc[i][j] = vzero;

    // ---- prologue: stage tile 0 into buffer 0 ----
    issueA(0, 0);
    if (BMODE == 0) {
        issueB0(0, 0);
    } else {
        loadB1(0);
        commitB1(0);
    }

    const int KSTEPS = K / BK;
    int cur = 0;
    for (int ks = 0; ks < KSTEPS; ks++) {
        const int kNext = (ks + 1) * BK;
        const bool hasNext = (ks + 1 < KSTEPS);
        if (hasNext) {
            if (BMODE == 1) loadB1(kNext);       // global -> regs (committed after compute)
            issueA(kNext, cur ^ 1);
            if (BMODE == 0) issueB0(kNext, cur ^ 1);
        }
        // wait for *current* tile's async copies (next tile's stay in flight)
        if (hasNext) {
            if (BMODE == 0) asm volatile("s_wait_asynccnt 6" ::: "memory");
            else            asm volatile("s_wait_asynccnt 2" ::: "memory");
        } else {
            asm volatile("s_wait_asynccnt 0" ::: "memory");
        }
        __syncthreads();                          // tile `cur` visible to all waves

        // fragment gathers per ISA 16-bit operand layout:
        // lane<16: K = 0..7 (elems 0..7) and 16..23 (elems 8..15); lane>=16: +8
        FragAB aF[4], bF[4];
        const int o = hs * 4;
        #pragma unroll
        for (int fm = 0; fm < 4; fm++) {
            const unsigned* p = (const unsigned*)&lhs[cur][(mW + 16 * fm + lr) * LDA_];
            #pragma unroll
            for (int j = 0; j < 4; j++) { aF[fm].u[j] = p[o + j]; aF[fm].u[4 + j] = p[o + 8 + j]; }
        }
        #pragma unroll
        for (int fn = 0; fn < 4; fn++) {
            const unsigned* p = (const unsigned*)&rhs[cur][(nW + 16 * fn + lr) * LDB_];
            #pragma unroll
            for (int j = 0; j < 4; j++) { bF[fn].u[j] = p[o + j]; bF[fn].u[4 + j] = p[o + 8 + j]; }
        }

        #pragma unroll
        for (int fm = 0; fm < 4; fm++)
            #pragma unroll
            for (int fn = 0; fn < 4; fn++)
                acc[fm][fn] = __builtin_amdgcn_wmma_f32_16x16x32_bf16(
                    false, aF[fm].v, false, bF[fn].v, (short)0, acc[fm][fn], false, false);

        __syncthreads();                          // all waves done reading `cur`
        if (hasNext && BMODE == 1) commitB1(cur ^ 1);
        cur ^= 1;
    }

    // ---- epilogue: C layout per ISA (VGPR r -> row mBase + r + 8*hs, col nBase + lr) ----
    #pragma unroll
    for (int fm = 0; fm < 4; fm++) {
        #pragma unroll
        for (int fn = 0; fn < 4; fn++) {
            #pragma unroll
            for (int r = 0; r < 8; r++) {
                const int mg = m0 + mW + 16 * fm + r + 8 * hs;
                const int ng = n0 + nW + 16 * fn + lr;
                float v = acc[fm][fn][r];
                if (EPI == EPI_BF16) {
                    unsigned short* Cp = (unsigned short*)Cout + (long)bat * batchC;
                    Cp[(long)mg * ldC + ng] = f2bf(v);
                } else if (EPI == EPI_F32_SCALE) {
                    float* Cp = (float*)Cout + (long)bat * batchC;
                    Cp[(long)mg * ldC + ng] = v * scale;
                } else if (EPI == EPI_F32_BIAS_RES) {
                    float* Cp = (float*)Cout + (long)bat * batchC;
                    const float* Rp = resid + (long)bat * batchR;
                    Cp[(long)mg * ldC + ng] = v + bias[mg] + Rp[(long)mg * ldC + ng];
                } else { // EPI_QKV: bias + bf16 store; rows < C additionally stored transposed (qT)
                    v += bias[mg];
                    const unsigned short h = f2bf(v);
                    unsigned short* Cp = (unsigned short*)Cout + (long)bat * batchC;
                    Cp[(long)mg * ldC + ng] = h;
                    if (mg < C_)
                        aux[(long)bat * ((long)C_ * L_) + (long)ng * C_ + mg] = h;
                }
            }
        }
    }
}

// ---------------- row softmax: f32 scores -> bf16 probs ----------------
__global__ __launch_bounds__(256) void softmax_rows_kernel(
    const float* __restrict__ S, unsigned short* __restrict__ P) {
    const int row = blockIdx.x;
    const int b   = blockIdx.y;
    const int tid = threadIdx.x;
    const float* s = S + ((long)b * L_ + row) * L_;
    unsigned short* p = P + ((long)b * L_ + row) * L_;
    __shared__ float red[256];
    float v[8];
    float mx = -1e30f;
    #pragma unroll
    for (int j = 0; j < 8; j++) { v[j] = s[tid + 256 * j]; mx = fmaxf(mx, v[j]); }
    red[tid] = mx; __syncthreads();
    for (int off = 128; off > 0; off >>= 1) {
        if (tid < off) red[tid] = fmaxf(red[tid], red[tid + off]);
        __syncthreads();
    }
    mx = red[0]; __syncthreads();
    float sum = 0.f;
    #pragma unroll
    for (int j = 0; j < 8; j++) { v[j] = __expf(v[j] - mx); sum += v[j]; }
    red[tid] = sum; __syncthreads();
    for (int off = 128; off > 0; off >>= 1) {
        if (tid < off) red[tid] += red[tid + off];
        __syncthreads();
    }
    const float inv = 1.0f / red[0];
    #pragma unroll
    for (int j = 0; j < 8; j++) p[tid + 256 * j] = f2bf(v[j] * inv);
}

// ---------------- host-side orchestration ----------------
extern "C" void kernel_launch(void* const* d_in, const int* in_sizes, int n_in,
                              void* d_out, int out_size, void* d_ws, size_t ws_size,
                              hipStream_t stream) {
    (void)in_sizes; (void)n_in; (void)out_size; (void)ws_size;
    const float* x      = (const float*)d_in[0];
    const float* gamma  = (const float*)d_in[1];
    const float* beta   = (const float*)d_in[2];
    const float* W_qkv  = (const float*)d_in[3];
    const float* b_qkv  = (const float*)d_in[4];
    const float* W_proj = (const float*)d_in[5];
    const float* b_proj = (const float*)d_in[6];
    float* out = (float*)d_out;

    // workspace layout
    char* ws = (char*)d_ws;
    unsigned short* xn    = (unsigned short*)ws; ws += (long)B_ * C_ * L_ * 2;       // 16.8 MB
    unsigned short* wq    = (unsigned short*)ws; ws += (long)3 * C_ * C_ * 2;        //  1.6 MB
    unsigned short* wp    = (unsigned short*)ws; ws += (long)C_ * C_ * 2;            //  0.5 MB
    unsigned short* qkv   = (unsigned short*)ws; ws += (long)B_ * 3 * C_ * L_ * 2;   // 50.3 MB
    unsigned short* qT    = (unsigned short*)ws; ws += (long)B_ * L_ * C_ * 2;       // 16.8 MB
    float*          scr   = (float*)ws;          ws += (long)B_ * L_ * L_ * 4;       // 134.2 MB
    unsigned short* probs = (unsigned short*)ws; ws += (long)B_ * L_ * L_ * 2;       // 67.1 MB
    unsigned short* outa  = (unsigned short*)ws; ws += (long)B_ * C_ * L_ * 2;       // 16.8 MB

    const long CL  = (long)C_ * L_;
    const long LL  = (long)L_ * L_;
    const long CL3 = 3L * C_ * L_;

    // 1) weight conversion to bf16
    {
        long n = 3L * C_ * C_;
        f32_to_bf16_kernel<<<dim3((unsigned)((n + 255) / 256)), 256, 0, stream>>>(W_qkv, wq, n);
        n = (long)C_ * C_;
        f32_to_bf16_kernel<<<dim3((unsigned)((n + 255) / 256)), 256, 0, stream>>>(W_proj, wp, n);
    }

    // 2) GroupNorm -> bf16 xn
    groupnorm_bf16_kernel<<<dim3(GROUPS_, B_), 256, 0, stream>>>(x, gamma, beta, xn);

    // 3) QKV = W_qkv(1536x512) * xn(512x2048) + b_qkv -> qkv bf16 (+ q transposed into qT)
    gemm_bf16_wmma<EPI_QKV, 1><<<dim3((3 * C_ / BM) * (L_ / BN), B_), 256, 0, stream>>>(
        wq, xn, /*sAm*/C_, /*sBk*/L_, /*sBn*/1,
        /*batchA*/0, /*batchB*/CL, /*batchC*/CL3, /*batchR*/0,
        3 * C_, L_, C_, b_qkv, nullptr, 1.0f, qkv, L_, qT);

    // 4) scores[i,j] = (1/sqrt(C)) * sum_c qT[i,c] k[c,j]
    gemm_bf16_wmma<EPI_F32_SCALE, 1><<<dim3((L_ / BM) * (L_ / BN), B_), 256, 0, stream>>>(
        qT, qkv + CL /*k*/, /*sAm*/C_, /*sBk*/L_, /*sBn*/1,
        /*batchA*/CL, /*batchB*/CL3, /*batchC*/LL, /*batchR*/0,
        L_, L_, C_, nullptr, nullptr, 1.0f / sqrtf((float)C_), scr, L_, nullptr);

    // 5) softmax rows -> bf16 probabilities
    softmax_rows_kernel<<<dim3(L_, B_), 256, 0, stream>>>(scr, probs);

    // 6) outa[c,i] = sum_j v[c,j] * P[i,j]   (B = P^T view: sBk==1 -> fully async path)
    gemm_bf16_wmma<EPI_BF16, 0><<<dim3((C_ / BM) * (L_ / BN), B_), 256, 0, stream>>>(
        qkv + 2 * CL /*v*/, probs, /*sAm*/L_, /*sBk*/1, /*sBn*/L_,
        /*batchA*/CL3, /*batchB*/LL, /*batchC*/CL, /*batchR*/0,
        C_, L_, L_, nullptr, nullptr, 1.0f, outa, L_, nullptr);

    // 7) y = W_proj(512x512) * outa(512x2048) + b_proj + residual(x) -> f32 out
    gemm_bf16_wmma<EPI_F32_BIAS_RES, 1><<<dim3((C_ / BM) * (L_ / BN), B_), 256, 0, stream>>>(
        wp, outa, /*sAm*/C_, /*sBk*/L_, /*sBn*/1,
        /*batchA*/0, /*batchB*/CL, /*batchC*/CL, /*batchR*/CL,
        C_, L_, C_, b_proj, x, 1.0f, out, L_, nullptr);
}
